// PointNet2TfModel_75282186764342
// MI455X (gfx1250) — matrix operand order repaired and checked by reading, not compiled, not run on hardware
//
#include <hip/hip_runtime.h>

typedef __attribute__((ext_vector_type(16))) __bf16 v16bf;
typedef __attribute__((ext_vector_type(8)))  __bf16 v8bf;
typedef __attribute__((ext_vector_type(8)))  float  v8f;

#define BATCH 8

// ---------------------------------------------------------------------------
// Farthest point sampling: one workgroup per batch, dist array in LDS.
// ---------------------------------------------------------------------------
__global__ __launch_bounds__(256) void k_fps(const float* __restrict__ xyz,
                                             int N, int npoint,
                                             int* __restrict__ fidx)
{
    __shared__ float dist[12288];
    __shared__ float rmax[256];
    __shared__ int   rarg[256];
    __shared__ int   cur;
    const int b   = blockIdx.x;
    const int tid = threadIdx.x;
    const float* X = xyz + (size_t)b * N * 3;
    if (tid == 0) cur = 0;
    for (int i = tid; i < N; i += 256) dist[i] = 1e10f;
    __syncthreads();
    for (int it = 0; it < npoint; ++it) {
        const int f = cur;
        if (tid == 0) fidx[b * npoint + it] = f;
        const float fx = X[f*3+0], fy = X[f*3+1], fz = X[f*3+2];
        float bm = -1.f; int ba = 0;
        for (int i = tid; i < N; i += 256) {
            float dx = X[i*3+0]-fx, dy = X[i*3+1]-fy, dz = X[i*3+2]-fz;
            float d  = dx*dx + dy*dy + dz*dz;
            float dd = fminf(dist[i], d);
            dist[i] = dd;
            if (dd > bm) { bm = dd; ba = i; }
        }
        rmax[tid] = bm; rarg[tid] = ba;
        __syncthreads();
        for (int s = 128; s > 0; s >>= 1) {
            if (tid < s && rmax[tid+s] > rmax[tid]) { rmax[tid] = rmax[tid+s]; rarg[tid] = rarg[tid+s]; }
            __syncthreads();
        }
        if (tid == 0) cur = rarg[0];
        __syncthreads();
    }
}

// ---------------------------------------------------------------------------
// Gather sampled centers.
// ---------------------------------------------------------------------------
__global__ void k_gather_xyz(const float* __restrict__ xyz, const int* __restrict__ idx,
                             float* __restrict__ out, int N, int S)
{
    int gid = blockIdx.x * 256 + threadIdx.x;
    if (gid >= BATCH * S) return;
    int b = gid / S, s = gid % S;
    int n = idx[b * S + s];
    const float* p = xyz + ((size_t)b * N + n) * 3;
    float* o = out + (size_t)gid * 3;
    o[0] = p[0]; o[1] = p[1]; o[2] = p[2];
}

// ---------------------------------------------------------------------------
// Ball-query grouping: one thread per group, scan N points in index order,
// keep first K inside radius; fill remainder with the first hit; writes
// bf16 rows [rel_xyz(3) | feats(C) | zero pad] of width Cpad.
// ---------------------------------------------------------------------------
__global__ void k_group(const float* __restrict__ xyz, const float* __restrict__ pts,
                        const float* __restrict__ nxyz, __bf16* __restrict__ A,
                        int N, int S, int K, int C, int Cpad, float r2)
{
    int gid = blockIdx.x * 256 + threadIdx.x;
    if (gid >= BATCH * S) return;
    int b = gid / S, s = gid % S;
    const float* ctr = nxyz + (size_t)gid * 3;
    const float cx = ctr[0], cy = ctr[1], cz = ctr[2];
    __bf16* out = A + (size_t)gid * K * Cpad;
    int cnt = 0;
    for (int n = 0; n < N && cnt < K; ++n) {
        const float* p = xyz + ((size_t)b * N + n) * 3;
        float dx = p[0]-cx, dy = p[1]-cy, dz = p[2]-cz;
        float d = dx*dx + dy*dy + dz*dz;
        if (d <= r2) {
            __bf16* o = out + (size_t)cnt * Cpad;
            o[0] = (__bf16)dx; o[1] = (__bf16)dy; o[2] = (__bf16)dz;
            const float* f = pts + ((size_t)b * N + n) * C;
            for (int c = 0; c < C; ++c)        o[3 + c] = (__bf16)f[c];
            for (int c = 3 + C; c < Cpad; ++c) o[c] = (__bf16)0.f;
            ++cnt;
        }
    }
    // replicate first hit into empty slots (slot 0 holds the first hit)
    for (; cnt < K; ++cnt) {
        __bf16* o = out + (size_t)cnt * Cpad;
        for (int c = 0; c < Cpad; ++c) o[c] = out[c];
    }
}

// ---------------------------------------------------------------------------
// Weight prep: f32 (C,F) row-major -> bf16 transposed+padded (F, Cpad).
// ---------------------------------------------------------------------------
__global__ void k_prep_w(const float* __restrict__ w, __bf16* __restrict__ wt,
                         int C, int F, int Cpad)
{
    int gid = blockIdx.x * 256 + threadIdx.x;
    if (gid >= F * Cpad) return;
    int f = gid / Cpad, c = gid % Cpad;
    wt[(size_t)f * Cpad + c] = (c < C) ? (__bf16)w[(size_t)c * F + f] : (__bf16)0.f;
}

// ---------------------------------------------------------------------------
// WMMA bf16 GEMM + ReLU:  Out[R,F] = relu(A[R,Cpad] * W[Cpad,F]).
// A bf16 row-major (stride Cpad); W pre-transposed bf16 [F, Cpad].
// One wave -> one 16x32 output tile (two accumulators sharing one A frag);
// K-loop in steps of 32. OUTF32 selects f32 (chain-final) vs bf16 output.
// ---------------------------------------------------------------------------
template <bool OUTF32>
__global__ __launch_bounds__(128) void k_gemm(const __bf16* __restrict__ A,
                                              const __bf16* __restrict__ Wt,
                                              void* __restrict__ Out,
                                              int R, int Cpad, int F)
{
    const int lane = threadIdx.x;                       // 0..31
    const int rt   = blockIdx.x * blockDim.y + threadIdx.y;
    if (rt * 16 >= R) return;
    const int row0 = rt * 16;
    const int col0 = blockIdx.y * 32;
    const int half = lane >> 4;
    const int l16  = lane & 15;

    v8f acc0 = {0.f,0.f,0.f,0.f,0.f,0.f,0.f,0.f};
    v8f acc1 = {0.f,0.f,0.f,0.f,0.f,0.f,0.f,0.f};

    const __bf16* arow  = A  + (size_t)(row0 + l16) * Cpad;       // A: M = lane&15
    const __bf16* brow0 = Wt + (size_t)(col0 + l16) * Cpad;       // B: N = lane&15
    const __bf16* brow1 = Wt + (size_t)(col0 + 16 + l16) * Cpad;

    for (int k0 = 0; k0 < Cpad; k0 += 32) {
        if (k0 + 32 < Cpad) __builtin_prefetch(arow + k0 + 32, 0, 0);
        // ---- A fragment: contiguous K16 chunk (k = 16*half + s), two 16B loads ----
        v8bf lo = *reinterpret_cast<const v8bf*>(arow + k0 + half * 16);
        v8bf hi = *reinterpret_cast<const v8bf*>(arow + k0 + half * 16 + 8);
        v16bf a = __builtin_shufflevector(lo, hi, 0,1,2,3,4,5,6,7,8,9,10,11,12,13,14,15);
        // ---- Two B fragments: contiguous K16 chunks of the transposed weight ----
        v16bf b0 = *reinterpret_cast<const v16bf*>(brow0 + k0 + half * 16);
        v16bf b1 = *reinterpret_cast<const v16bf*>(brow1 + k0 + half * 16);
        acc0 = __builtin_amdgcn_wmma_f32_16x16x32_bf16(
                 false, a, false, b0, (short)0, acc0, false, false);
        acc1 = __builtin_amdgcn_wmma_f32_16x16x32_bf16(
                 false, a, false, b1, (short)0, acc1, false, false);
    }

    // D layout: VGPR r -> row r + 8*(lane>>4); col = lane&15. Fused ReLU.
    const size_t orow = (size_t)(row0 + half * 8);
    if constexpr (OUTF32) {
        float* O = (float*)Out;
        #pragma unroll
        for (int r = 0; r < 8; ++r) {
            O[(orow + r) * F + col0 + l16]      = fmaxf(acc0[r], 0.f);
            O[(orow + r) * F + col0 + 16 + l16] = fmaxf(acc1[r], 0.f);
        }
    } else {
        __bf16* O = (__bf16*)Out;
        #pragma unroll
        for (int r = 0; r < 8; ++r) {
            O[(orow + r) * F + col0 + l16]      = (__bf16)fmaxf(acc0[r], 0.f);
            O[(orow + r) * F + col0 + 16 + l16] = (__bf16)fmaxf(acc1[r], 0.f);
        }
    }
}

// ---------------------------------------------------------------------------
// Max-pool over the K (nsample) axis (f32 in from chain-final GEMM).
// ---------------------------------------------------------------------------
__global__ void k_maxpool(const float* __restrict__ in, float* __restrict__ out,
                          int G, int K, int F)
{
    int gid = blockIdx.x * 256 + threadIdx.x;
    if (gid >= G * F) return;
    int g = gid / F, f = gid % F;
    const float* p = in + (size_t)g * K * F + f;
    float m = -1e30f;
    for (int k = 0; k < K; ++k) m = fmaxf(m, p[(size_t)k * F]);
    out[(size_t)g * F + f] = m;
}

// ---------------------------------------------------------------------------
// FP stage: 3-NN inverse-distance interpolation + concat [interp | target],
// zero-padded to Cpad; emits the bf16 GEMM A matrix directly.
// ---------------------------------------------------------------------------
__global__ void k_interp_concat(const float* __restrict__ txyz, const float* __restrict__ sxyz,
                                const float* __restrict__ tfeat, const float* __restrict__ sfeat,
                                __bf16* __restrict__ A,
                                int Nt, int Ns, int Ct, int Cs, int Cpad)
{
    int gid = blockIdx.x * 256 + threadIdx.x;
    if (gid >= BATCH * Nt) return;
    int b = gid / Nt;
    const float* t = txyz + (size_t)gid * 3;
    const float tx = t[0], ty = t[1], tz = t[2];
    float d0 = 1e30f, d1 = 1e30f, d2 = 1e30f;
    int   i0 = 0,     i1 = 0,     i2 = 0;
    const float* S = sxyz + (size_t)b * Ns * 3;
    for (int s = 0; s < Ns; ++s) {
        float dx = S[s*3+0]-tx, dy = S[s*3+1]-ty, dz = S[s*3+2]-tz;
        float d  = sqrtf(fmaxf(dx*dx + dy*dy + dz*dz, 0.f));
        if (d < d0)      { d2=d1; i2=i1; d1=d0; i1=i0; d0=d; i0=s; }
        else if (d < d1) { d2=d1; i2=i1; d1=d;  i1=s; }
        else if (d < d2) { d2=d;  i2=s; }
    }
    d0 = fmaxf(d0, 1e-10f); d1 = fmaxf(d1, 1e-10f); d2 = fmaxf(d2, 1e-10f);
    float r0 = 1.f/d0, r1 = 1.f/d1, r2 = 1.f/d2;
    float nrm = r0 + r1 + r2;
    float w0 = r0/nrm, w1 = r1/nrm, w2 = r2/nrm;
    float wsum = w0 + w1 + w2 + 1e-6f;
    const float* F0 = sfeat + ((size_t)b * Ns + i0) * Cs;
    const float* F1 = sfeat + ((size_t)b * Ns + i1) * Cs;
    const float* F2 = sfeat + ((size_t)b * Ns + i2) * Cs;
    __bf16* out = A + (size_t)gid * Cpad;
    for (int c = 0; c < Cs; ++c)
        out[c] = (__bf16)((w0 * F0[c] + w1 * F1[c] + w2 * F2[c]) / wsum);
    const float* TF = tfeat + (size_t)gid * Ct;
    for (int c = 0; c < Ct; ++c) out[Cs + c] = (__bf16)TF[c];
    for (int c = Cs + Ct; c < Cpad; ++c) out[c] = (__bf16)0.f;
}

// ---------------------------------------------------------------------------
// Host orchestration.
// ---------------------------------------------------------------------------
extern "C" void kernel_launch(void* const* d_in, const int* in_sizes, int n_in,
                              void* d_out, int out_size, void* d_ws, size_t ws_size,
                              hipStream_t stream)
{
    (void)in_sizes; (void)n_in; (void)out_size; (void)ws_size;
    const float* xyz0 = (const float*)d_in[0];   // [8,12288,3]
    const float* pts0 = (const float*)d_in[1];   // [8,12288,6]
    const float* w[21];
    for (int i = 0; i < 21; ++i) w[i] = (const float*)d_in[2 + i];

    // weight dims: (C, F, Cpad) in d_in order
    static const int WC[21]  = {  9, 32, 32,  67, 64, 64, 131,128,128, 259,256,256, 768,256, 384,256, 320,256, 134,128,128 };
    static const int WF[21]  = { 32, 32, 64,  64, 64,128, 128,128,256, 256,256,512, 256,256, 256,256, 256,128, 128,128,128 };
    static const int WCP[21] = { 32, 32, 32,  96, 64, 64, 160,128,128, 288,256,256, 768,256, 384,256, 320,256, 160,128,128 };

    char* base = (char*)d_ws; size_t off = 0;
    auto alloc = [&](size_t bytes) -> void* {
        void* p = base + off; off = (off + bytes + 255) & ~(size_t)255; return p;
    };
    auto g1 = [](long n) { return dim3((unsigned)((n + 255) / 256)); };

    // --- workspace layout ---
    __bf16* wt[21];
    for (int i = 0; i < 21; ++i) wt[i] = (__bf16*)alloc((size_t)2 * WF[i] * WCP[i]);
    int*   fidx1 = (int*)alloc(4u * BATCH * 512);
    int*   fidx2 = (int*)alloc(4u * BATCH * 128);
    int*   fidx3 = (int*)alloc(4u * BATCH * 32);
    int*   fidx4 = (int*)alloc(4u * BATCH * 8);
    float* xyz1  = (float*)alloc(4ull * BATCH * 512 * 3);
    float* xyz2  = (float*)alloc(4ull * BATCH * 128 * 3);
    float* xyz3  = (float*)alloc(4ull * BATCH * 32 * 3);
    float* xyz4  = (float*)alloc(4ull * BATCH * 8 * 3);
    float* l1pts = (float*)alloc(4ull * BATCH * 512 * 64);
    float* l2pts = (float*)alloc(4ull * BATCH * 128 * 128);
    float* l3pts = (float*)alloc(4ull * BATCH * 32 * 256);
    float* l4pts = (float*)alloc(4ull * BATCH * 8 * 512);
    float* l3p   = (float*)alloc(4ull * BATCH * 32 * 256);
    float* l2p   = (float*)alloc(4ull * BATCH * 128 * 256);
    float* l1p   = (float*)alloc(4ull * BATCH * 512 * 128);
    __bf16* bufA = (__bf16*)alloc(2ull * 98304 * 160);  // largest GEMM A (bf16)
    void*  bufB  = alloc(4ull * 131072 * 64);           // ping (bf16 or f32)
    void*  bufC  = alloc(4ull * 131072 * 64);           // pong

    for (int i = 0; i < 21; ++i)
        k_prep_w<<<g1((long)WF[i] * WCP[i]), 256, 0, stream>>>(w[i], wt[i], WC[i], WF[i], WCP[i]);

    auto gemm = [&](const __bf16* A, const __bf16* Wt, void* Out, int R, int Cp, int F, bool outf32) {
        dim3 blk(32, 4), grd((unsigned)((R / 16 + 3) / 4), (unsigned)(F / 32));
        if (outf32) k_gemm<true ><<<grd, blk, 0, stream>>>(A, Wt, Out, R, Cp, F);
        else        k_gemm<false><<<grd, blk, 0, stream>>>(A, Wt, Out, R, Cp, F);
    };

    // ---------------- SA1: N=12288 -> S=512, r=0.1, K=32, C=6, Cpad=32 ----------------
    k_fps<<<BATCH, 256, 0, stream>>>(xyz0, 12288, 512, fidx1);
    k_gather_xyz<<<g1(BATCH * 512), 256, 0, stream>>>(xyz0, fidx1, xyz1, 12288, 512);
    k_group<<<g1(BATCH * 512), 256, 0, stream>>>(xyz0, pts0, xyz1, bufA, 12288, 512, 32, 6, 32, 0.01f);
    gemm(bufA, wt[0], bufB, BATCH*512*32, 32, 32, false);
    gemm((const __bf16*)bufB, wt[1], bufC, BATCH*512*32, 32, 32, false);
    gemm((const __bf16*)bufC, wt[2], bufB, BATCH*512*32, 32, 64, true);
    k_maxpool<<<g1((long)BATCH*512*64), 256, 0, stream>>>((const float*)bufB, l1pts, BATCH*512, 32, 64);

    // ---------------- SA2: 512 -> 128, r=0.2, C=64, Cpad=96 ----------------
    k_fps<<<BATCH, 256, 0, stream>>>(xyz1, 512, 128, fidx2);
    k_gather_xyz<<<g1(BATCH * 128), 256, 0, stream>>>(xyz1, fidx2, xyz2, 512, 128);
    k_group<<<g1(BATCH * 128), 256, 0, stream>>>(xyz1, l1pts, xyz2, bufA, 512, 128, 32, 64, 96, 0.04f);
    gemm(bufA, wt[3], bufB, BATCH*128*32, 96, 64, false);
    gemm((const __bf16*)bufB, wt[4], bufC, BATCH*128*32, 64, 64, false);
    gemm((const __bf16*)bufC, wt[5], bufB, BATCH*128*32, 64, 128, true);
    k_maxpool<<<g1((long)BATCH*128*128), 256, 0, stream>>>((const float*)bufB, l2pts, BATCH*128, 32, 128);

    // ---------------- SA3: 128 -> 32, r=0.4, C=128, Cpad=160 ----------------
    k_fps<<<BATCH, 256, 0, stream>>>(xyz2, 128, 32, fidx3);
    k_gather_xyz<<<g1(BATCH * 32), 256, 0, stream>>>(xyz2, fidx3, xyz3, 128, 32);
    k_group<<<g1(BATCH * 32), 256, 0, stream>>>(xyz2, l2pts, xyz3, bufA, 128, 32, 32, 128, 160, 0.16f);
    gemm(bufA, wt[6], bufB, BATCH*32*32, 160, 128, false);
    gemm((const __bf16*)bufB, wt[7], bufC, BATCH*32*32, 128, 128, false);
    gemm((const __bf16*)bufC, wt[8], bufB, BATCH*32*32, 128, 256, true);
    k_maxpool<<<g1((long)BATCH*32*256), 256, 0, stream>>>((const float*)bufB, l3pts, BATCH*32, 32, 256);

    // ---------------- SA4: 32 -> 8, r=0.8, C=256, Cpad=288 ----------------
    k_fps<<<BATCH, 256, 0, stream>>>(xyz3, 32, 8, fidx4);
    k_gather_xyz<<<g1(BATCH * 8), 256, 0, stream>>>(xyz3, fidx4, xyz4, 32, 8);
    k_group<<<g1(BATCH * 8), 256, 0, stream>>>(xyz3, l3pts, xyz4, bufA, 32, 8, 32, 256, 288, 0.64f);
    gemm(bufA, wt[9],  bufB, BATCH*8*32, 288, 256, false);
    gemm((const __bf16*)bufB, wt[10], bufC, BATCH*8*32, 256, 256, false);
    gemm((const __bf16*)bufC, wt[11], bufB, BATCH*8*32, 256, 512, true);
    k_maxpool<<<g1((long)BATCH*8*512), 256, 0, stream>>>((const float*)bufB, l4pts, BATCH*8, 32, 512);

    // ---------------- FP1: l4 -> l3 (Cs=512, Ct=256, Cpad=768) ----------------
    k_interp_concat<<<g1(BATCH * 32), 256, 0, stream>>>(xyz3, xyz4, l3pts, l4pts, bufA, 32, 8, 256, 512, 768);
    gemm(bufA, wt[12], bufB, BATCH*32, 768, 256, false);
    gemm((const __bf16*)bufB, wt[13], l3p, BATCH*32, 256, 256, true);

    // ---------------- FP2: l3 -> l2 (Cs=256, Ct=128, Cpad=384) ----------------
    k_interp_concat<<<g1(BATCH * 128), 256, 0, stream>>>(xyz2, xyz3, l2pts, l3p, bufA, 128, 32, 128, 256, 384);
    gemm(bufA, wt[14], bufB, BATCH*128, 384, 256, false);
    gemm((const __bf16*)bufB, wt[15], l2p, BATCH*128, 256, 256, true);

    // ---------------- FP3: l2 -> l1 (Cs=256, Ct=64, Cpad=320) ----------------
    k_interp_concat<<<g1(BATCH * 512), 256, 0, stream>>>(xyz1, xyz2, l1pts, l2p, bufA, 512, 128, 64, 256, 320);
    gemm(bufA, wt[16], bufB, BATCH*512, 320, 256, false);
    gemm((const __bf16*)bufB, wt[17], l1p, BATCH*512, 256, 128, true);

    // ---------------- FP4: l1 -> l0 (Cs=128, Ct=6, Cpad=160), final -> d_out ----------------
    k_interp_concat<<<g1(BATCH * 12288), 256, 0, stream>>>(xyz0, xyz1, pts0, l1p, bufA, 12288, 512, 6, 128, 160);
    gemm(bufA, wt[18], bufB, BATCH*12288, 160, 128, false);
    gemm((const __bf16*)bufB, wt[19], bufC, BATCH*12288, 128, 128, false);
    gemm((const __bf16*)bufC, wt[20], (float*)d_out, BATCH*12288, 128, 128, true);
}
